// VisionTransformerForSegmentation_69380901699767
// MI455X (gfx1250) — matrix operand order repaired and checked by reading, not compiled, not run on hardware
//
#include <hip/hip_runtime.h>
#include <hip/hip_bf16.h>
#include <math.h>

typedef __bf16 bf16;
typedef __attribute__((ext_vector_type(16))) __bf16 v16bf;
typedef __attribute__((ext_vector_type(8)))  float   v8f;

#define IMG   512
#define PP    16
#define CIN   3
#define COUTC 3
#define EMB   768
#define NHEAD 12
#define LAY   8
#define NPH   32
#define NT    1024
#define HD    64
#define FF    1536
#define TE    2304
#define INTR  128
#define BATCH 2
#define MROWS 2048   // BATCH*NT
#define BH    24     // BATCH*NHEAD

// ---------------- bf16 <-> f32 helpers (bit-exact RNE, no libdevice dependency) ------------
__device__ __forceinline__ bf16 f2b(float f) {
  unsigned u = __float_as_uint(f);
  u += 0x7fffu + ((u >> 16) & 1u);
  unsigned short s = (unsigned short)(u >> 16);
  bf16 r; __builtin_memcpy(&r, &s, 2); return r;
}
__device__ __forceinline__ float b2f(bf16 b) {
  unsigned short s; __builtin_memcpy(&s, &b, 2);
  return __uint_as_float(((unsigned)s) << 16);
}

// ---------------- WMMA fragment loaders (ISA 7.12.2 layouts, wave32) -----------------------
// A: 16x32 (MxK). lanes 0-15: row=lane, K={0..7,16..23}; lanes 16-31: row=lane-16, K={8..15,24..31}
__device__ __forceinline__ v16bf load_a(const bf16* __restrict__ A, int lda) {
  const int lane = threadIdx.x & 31;
  const bf16* p = A + (size_t)(lane & 15) * lda + ((lane >> 4) << 3);
  v16bf a;
#pragma unroll
  for (int i = 0; i < 8; ++i) { a[i] = p[i]; a[8 + i] = p[16 + i]; }
  return a;
}
__device__ __forceinline__ v16bf load_a_f32(const float* __restrict__ A, int lda) {
  const int lane = threadIdx.x & 31;
  const float* p = A + (size_t)(lane & 15) * lda + ((lane >> 4) << 3);
  v16bf a;
#pragma unroll
  for (int i = 0; i < 8; ++i) { a[i] = f2b(p[i]); a[8 + i] = f2b(p[16 + i]); }
  return a;
}
// B from row-major W (Nc x K): C = A * W^T.  lane col=lane&15, K = 8*(lane>>4)+{0..7,16..23}, same as A.
__device__ __forceinline__ v16bf load_bT(const bf16* __restrict__ W, int ldw) {
  const int lane = threadIdx.x & 31;
  const bf16* p = W + (size_t)(lane & 15) * ldw + ((lane >> 4) << 3);
  v16bf b;
#pragma unroll
  for (int i = 0; i < 8; ++i) { b[i] = p[i]; b[8 + i] = p[16 + i]; }
  return b;
}

__device__ __forceinline__ v8f wmma_bf16(v16bf a, v16bf b, v8f c) {
  return __builtin_amdgcn_wmma_f32_16x16x32_bf16(false, a, false, b, (short)0, c, false, false);
}

// 16x64 output tile: one A fragment reused across 4 B fragments (4 accumulators).
__device__ __forceinline__ void tile_gemm4(const bf16* __restrict__ A, int lda,
                                           const bf16* __restrict__ W, int ldw, int K,
                                           v8f acc[4]) {
  for (int k = 0; k < K; k += 32) {
    v16bf a = load_a(A + k, lda);
#pragma unroll
    for (int s = 0; s < 4; ++s)
      acc[s] = wmma_bf16(a, load_bT(W + (size_t)(s * 16) * ldw + k, ldw), acc[s]);
  }
}

// ---------------- misc elementwise kernels -------------------------------------------------
__global__ void k_cvt(const float* __restrict__ s, bf16* __restrict__ d, int n) {
  int i = blockIdx.x * 256 + threadIdx.x;
  if (i < n) d[i] = f2b(s[i]);
}

__global__ void k_pe(float* __restrict__ pe) {
  int idx = blockIdx.x * 256 + threadIdx.x;            // NT*EMB
  if (idx >= NT * EMB) return;
  int d = idx % EMB, n = idx / EMB;
  int r = n >> 5, c = n & 31;
  const int dq = EMB / 4, dh = EMB / 2;                // 192, 384
  float v = 0.f;
  if (d < dh) {
    int dd = (d < dq) ? d : (d - dq);
    int jj = dd >> 1;
    float div = expf((float)(2 * jj) * (-logf(10000.f) / (float)dq));
    float pos = (d < dq) ? (float)r : (float)c;
    v = (dd & 1) ? cosf(pos * div) : sinf(pos * div);
  }
  pe[idx] = v;
}

__global__ void k_im2col(const float* __restrict__ x, bf16* __restrict__ xp) {
  int idx = blockIdx.x * 256 + threadIdx.x;            // BATCH*NT*EMB
  if (idx >= BATCH * NT * EMB) return;
  int col = idx % EMB, n = (idx / EMB) % NT, b = idx / (EMB * NT);
  int c = col / (PP * PP), py = (col / PP) % PP, px = col % PP;
  int r = n >> 5, cc = n & 31;
  xp[idx] = f2b(x[(((size_t)b * CIN + c) * IMG + (r * PP + py)) * IMG + (cc * PP + px)]);
}

__global__ void k_ln(const float* __restrict__ x, const float* __restrict__ g,
                     const float* __restrict__ b, bf16* __restrict__ y) {
  int r = blockIdx.x;
  const float* row = x + (size_t)r * EMB;
  __shared__ float red[256];
  int t = threadIdx.x;
  float s = 0.f;
  for (int i = t; i < EMB; i += 256) s += row[i];
  red[t] = s; __syncthreads();
  for (int st = 128; st > 0; st >>= 1) { if (t < st) red[t] += red[t + st]; __syncthreads(); }
  float mean = red[0] / (float)EMB; __syncthreads();
  float v = 0.f;
  for (int i = t; i < EMB; i += 256) { float d = row[i] - mean; v += d * d; }
  red[t] = v; __syncthreads();
  for (int st = 128; st > 0; st >>= 1) { if (t < st) red[t] += red[t + st]; __syncthreads(); }
  float inv = rsqrtf(red[0] / (float)EMB + 1e-5f);
  for (int i = t; i < EMB; i += 256)
    y[(size_t)r * EMB + i] = f2b((row[i] - mean) * inv * g[i] + b[i]);
}

__global__ void k_softmax(float* __restrict__ S) {
  int n = blockIdx.x, z = blockIdx.y;
  float* row = S + ((size_t)z * NT + n) * NT;
  __shared__ float red[256];
  int t = threadIdx.x;
  float m = -1e30f;
  for (int i = t; i < NT; i += 256) m = fmaxf(m, row[i]);
  red[t] = m; __syncthreads();
  for (int s = 128; s > 0; s >>= 1) { if (t < s) red[t] = fmaxf(red[t], red[t + s]); __syncthreads(); }
  m = red[0]; __syncthreads();
  float sum = 0.f;
  for (int i = t; i < NT; i += 256) { float v = expf(row[i] - m); row[i] = v; sum += v; }
  red[t] = sum; __syncthreads();
  for (int s = 128; s > 0; s >>= 1) { if (t < s) red[t] += red[t + s]; __syncthreads(); }
  float inv = 1.f / red[0];
  for (int i = t; i < NT; i += 256) row[i] *= inv;
}

// ---------------- GEMM kernels (1 wave = one 16x64 f32 tile) -------------------------------
__global__ void k_patch_gemm(const bf16* __restrict__ xp, const bf16* __restrict__ W,
                             const float* __restrict__ bias, const float* __restrict__ pe,
                             float* __restrict__ t) {
  int tm = blockIdx.x * 16, tn = blockIdx.y * 64;
  v8f acc[4] = {};
  tile_gemm4(xp + (size_t)tm * EMB, EMB, W + (size_t)tn * EMB, EMB, EMB, acc);
  int lane = threadIdx.x & 31;
  int row0 = tm + ((lane >> 4) << 3);
#pragma unroll
  for (int s = 0; s < 4; ++s) {
    int col = tn + s * 16 + (lane & 15);
#pragma unroll
    for (int e = 0; e < 8; ++e) {
      int r = row0 + e, n = r & (NT - 1);
      t[(size_t)r * EMB + col] = acc[s][e] + bias[col] + pe[(size_t)n * EMB + col];
    }
  }
}

__global__ void k_qkv_gemm(const bf16* __restrict__ y, const bf16* __restrict__ W,
                           bf16* __restrict__ q, bf16* __restrict__ kk, bf16* __restrict__ vt) {
  int tm = blockIdx.x * 16, tn = blockIdx.y * 64;
  v8f acc[4] = {};
  tile_gemm4(y + (size_t)tm * EMB, EMB, W + (size_t)tn * EMB, EMB, EMB, acc);
  int lane = threadIdx.x & 31;
  int row0 = tm + ((lane >> 4) << 3);
#pragma unroll
  for (int s = 0; s < 4; ++s) {
    int j = tn + s * 16 + (lane & 15);
    int which = j / EMB, jj = j % EMB, h = jj / HD, d = jj % HD;
#pragma unroll
    for (int e = 0; e < 8; ++e) {
      int r = row0 + e, b = r >> 10, n = r & (NT - 1);
      size_t bh = (size_t)(b * NHEAD + h);
      bf16 val = f2b(acc[s][e]);
      if (which == 0)      q [(bh * NT + n) * HD + d] = val;
      else if (which == 1) kk[(bh * NT + n) * HD + d] = val;
      else                 vt[(bh * HD + d) * NT + n] = val;    // V stored transposed
    }
  }
}

__global__ void k_score_gemm(const bf16* __restrict__ q, const bf16* __restrict__ kk,
                             const float* __restrict__ bias, float* __restrict__ S) {
  int z = blockIdx.z;
  int tm = blockIdx.x * 16, tn = blockIdx.y * 64;
  v8f acc[4] = {};
  tile_gemm4(q + (size_t)z * NT * HD + (size_t)tm * HD, HD,
             kk + (size_t)z * NT * HD + (size_t)tn * HD, HD, HD, acc);
  int lane = threadIdx.x & 31;
  int row0 = tm + ((lane >> 4) << 3);
  float* Sz = S + (size_t)z * NT * NT;
#pragma unroll
  for (int s = 0; s < 4; ++s) {
    int col = tn + s * 16 + (lane & 15);
#pragma unroll
    for (int e = 0; e < 8; ++e) {
      int r = row0 + e;
      Sz[(size_t)r * NT + col] = acc[s][e] * 0.125f + bias[(size_t)r * NT + col];
    }
  }
}

__global__ void k_av_gemm(const float* __restrict__ S, const bf16* __restrict__ vt,
                          bf16* __restrict__ o) {
  int z = blockIdx.z;
  int tm = blockIdx.x * 16;                                // all 64 head-dims per wave
  const float* A = S + (size_t)z * NT * NT + (size_t)tm * NT;
  const bf16* W = vt + (size_t)z * HD * NT;
  v8f acc[4] = {};
  for (int k = 0; k < NT; k += 32) {
    v16bf a = load_a_f32(A + k, NT);                       // probs f32 -> bf16, amortized 4x
#pragma unroll
    for (int s = 0; s < 4; ++s)
      acc[s] = wmma_bf16(a, load_bT(W + (size_t)(s * 16) * NT + k, NT), acc[s]);
  }
  int lane = threadIdx.x & 31;
  int row0 = tm + ((lane >> 4) << 3);
  int b = z / NHEAD, h = z % NHEAD;
#pragma unroll
  for (int s = 0; s < 4; ++s) {
    int d = s * 16 + (lane & 15);
#pragma unroll
    for (int e = 0; e < 8; ++e) {
      int n = row0 + e;
      o[((size_t)(b * NT + n)) * EMB + h * HD + d] = f2b(acc[s][e]);
    }
  }
}

__global__ void k_proj_gemm(const bf16* __restrict__ o, const bf16* __restrict__ W,
                            const float* __restrict__ pb, const bf16* __restrict__ y,
                            float* __restrict__ t) {
  int tm = blockIdx.x * 16, tn = blockIdx.y * 64;
  v8f acc[4] = {};
  tile_gemm4(o + (size_t)tm * EMB, EMB, W + (size_t)tn * EMB, EMB, EMB, acc);
  int lane = threadIdx.x & 31;
  int row0 = tm + ((lane >> 4) << 3);
#pragma unroll
  for (int s = 0; s < 4; ++s) {
    int col = tn + s * 16 + (lane & 15);
#pragma unroll
    for (int e = 0; e < 8; ++e) {
      size_t idx = (size_t)(row0 + e) * EMB + col;
      t[idx] += b2f(y[idx]) + acc[s][e] + pb[col];          // t += y + o (o incl. bias)
    }
  }
}

__global__ void k_mlp1_gemm(const bf16* __restrict__ h, const bf16* __restrict__ W,
                            const float* __restrict__ b1, bf16* __restrict__ h1) {
  int tm = blockIdx.x * 16, tn = blockIdx.y * 64;
  v8f acc[4] = {};
  tile_gemm4(h + (size_t)tm * EMB, EMB, W + (size_t)tn * EMB, EMB, EMB, acc);
  int lane = threadIdx.x & 31;
  int row0 = tm + ((lane >> 4) << 3);
#pragma unroll
  for (int s = 0; s < 4; ++s) {
    int col = tn + s * 16 + (lane & 15);
#pragma unroll
    for (int e = 0; e < 8; ++e) {
      float xg = acc[s][e] + b1[col];
      xg = 0.5f * xg * (1.f + erff(xg * 0.70710678118f));   // exact GELU
      h1[(size_t)(row0 + e) * FF + col] = f2b(xg);
    }
  }
}

__global__ void k_mlp2_gemm(const bf16* __restrict__ h1, const bf16* __restrict__ W,
                            const float* __restrict__ b2, float* __restrict__ t) {
  int tm = blockIdx.x * 16, tn = blockIdx.y * 64;
  v8f acc[4] = {};
  tile_gemm4(h1 + (size_t)tm * FF, FF, W + (size_t)tn * FF, FF, FF, acc);
  int lane = threadIdx.x & 31;
  int row0 = tm + ((lane >> 4) << 3);
#pragma unroll
  for (int s = 0; s < 4; ++s) {
    int col = tn + s * 16 + (lane & 15);
#pragma unroll
    for (int e = 0; e < 8; ++e) {
      size_t idx = (size_t)(row0 + e) * EMB + col;
      t[idx] += acc[s][e] + b2[col];
    }
  }
}

__global__ void k_op1_gemm(const bf16* __restrict__ tb, const bf16* __restrict__ W,
                           const float* __restrict__ b1, bf16* __restrict__ oh) {
  int tm = blockIdx.x * 16, tn = blockIdx.y * 64;
  v8f acc[4] = {};
  tile_gemm4(tb + (size_t)tm * EMB, EMB, W + (size_t)tn * EMB, EMB, EMB, acc);
  int lane = threadIdx.x & 31;
  int row0 = tm + ((lane >> 4) << 3);
#pragma unroll
  for (int s = 0; s < 4; ++s) {
    int col = tn + s * 16 + (lane & 15);
#pragma unroll
    for (int e = 0; e < 8; ++e)
      oh[(size_t)(row0 + e) * INTR + col] = f2b(fmaxf(acc[s][e] + b1[col], 0.f));
  }
}

__global__ void k_op2_gemm(const bf16* __restrict__ oh, const bf16* __restrict__ W,
                           const float* __restrict__ b2, float* __restrict__ h2) {
  int tm = blockIdx.x * 16, tn = blockIdx.y * 64;
  v8f acc[4] = {};
  tile_gemm4(oh + (size_t)tm * INTR, INTR, W + (size_t)tn * INTR, INTR, INTR, acc);
  int lane = threadIdx.x & 31;
  int row0 = tm + ((lane >> 4) << 3);
#pragma unroll
  for (int s = 0; s < 4; ++s) {
    int col = tn + s * 16 + (lane & 15);
#pragma unroll
    for (int e = 0; e < 8; ++e)
      h2[(size_t)(row0 + e) * EMB + col] = acc[s][e] + b2[col];
  }
}

// ---------------- fold + attention-gate ----------------------------------------------------
__global__ void k_gate(const float* __restrict__ h2, const float* __restrict__ x,
                       const float* __restrict__ wg, const float* __restrict__ bngw,
                       const float* __restrict__ bngb, const float* __restrict__ wx,
                       const float* __restrict__ bnxw, const float* __restrict__ bnxb,
                       const float* __restrict__ psiw, const float* __restrict__ psib,
                       float* __restrict__ out) {
  int idx = blockIdx.x * 256 + threadIdx.x;                 // BATCH*IMG*IMG
  if (idx >= BATCH * IMG * IMG) return;
  int ww = idx % IMG, hh = (idx / IMG) % IMG, b = idx / (IMG * IMG);
  int r = hh >> 4, py = hh & 15, cc = ww >> 4, px = ww & 15;
  int n = r * NPH + cc;
  float inv = rsqrtf(1.f + 1e-5f);                          // BN eval, mean=0 var=1
  float gch[COUTC], xch[CIN], gs = 0.f, xs = 0.f;
#pragma unroll
  for (int c = 0; c < COUTC; ++c) {
    float g = h2[((size_t)(b * NT + n)) * EMB + c * PP * PP + py * PP + px];
    gch[c] = g; gs += g * wg[c];
  }
#pragma unroll
  for (int c = 0; c < CIN; ++c) {
    float xv = x[(((size_t)b * CIN + c) * IMG + hh) * IMG + ww];
    xch[c] = xv; xs += xv * wx[c];
  }
  float g1 = gs * inv * bngw[0] + bngb[0];
  float x1 = xs * inv * bnxw[0] + bnxb[0];
  float rl = fmaxf(g1 + x1, 0.f);
  float psi = 1.f / (1.f + expf(-(rl * psiw[0] + psib[0])));
#pragma unroll
  for (int c = 0; c < COUTC; ++c)
    out[(((size_t)b * COUTC + c) * IMG + hh) * IMG + ww] = gch[c] * xch[c] * psi;
}

// ---------------- host orchestration -------------------------------------------------------
extern "C" void kernel_launch(void* const* d_in, const int* in_sizes, int n_in,
                              void* d_out, int out_size, void* d_ws, size_t ws_size,
                              hipStream_t stream) {
  const float* x       = (const float*)d_in[0];
  const float* conv_w  = (const float*)d_in[1];
  const float* conv_b  = (const float*)d_in[2];
  const float* ln1_g   = (const float*)d_in[3];
  const float* ln1_b   = (const float*)d_in[4];
  const float* qkv_w   = (const float*)d_in[5];
  const float* proj_w  = (const float*)d_in[6];
  const float* proj_b  = (const float*)d_in[7];
  const float* attn_bi = (const float*)d_in[8];
  const float* ln2_g   = (const float*)d_in[9];
  const float* ln2_b   = (const float*)d_in[10];
  const float* mlp_w1  = (const float*)d_in[11];
  const float* mlp_b1  = (const float*)d_in[12];
  const float* mlp_w2  = (const float*)d_in[13];
  const float* mlp_b2  = (const float*)d_in[14];
  const float* op_w1   = (const float*)d_in[15];
  const float* op_b1   = (const float*)d_in[16];
  const float* op_w2   = (const float*)d_in[17];
  const float* op_b2   = (const float*)d_in[18];
  const float* ag_wg   = (const float*)d_in[19];
  const float* ag_bngw = (const float*)d_in[20];
  const float* ag_bngb = (const float*)d_in[21];
  const float* ag_wx   = (const float*)d_in[22];
  const float* ag_bnxw = (const float*)d_in[23];
  const float* ag_bnxb = (const float*)d_in[24];
  const float* ag_psiw = (const float*)d_in[25];
  const float* ag_psib = (const float*)d_in[26];
  float* out = (float*)d_out;

  char* base = (char*)d_ws; size_t off = 0;
  auto take = [&](size_t bytes) -> void* {
    off = (off + 255) & ~(size_t)255;
    void* p = base + off; off += bytes; return p;
  };
  bf16*  wb_conv = (bf16*) take((size_t)EMB * EMB * 2);
  bf16*  wb_qkv  = (bf16*) take((size_t)LAY * TE * EMB * 2);
  bf16*  wb_proj = (bf16*) take((size_t)LAY * EMB * EMB * 2);
  bf16*  wb_m1   = (bf16*) take((size_t)LAY * FF * EMB * 2);
  bf16*  wb_m2   = (bf16*) take((size_t)LAY * EMB * FF * 2);
  bf16*  wb_o1   = (bf16*) take((size_t)INTR * EMB * 2);
  bf16*  wb_o2   = (bf16*) take((size_t)EMB * INTR * 2);
  float* pe      = (float*)take((size_t)NT * EMB * 4);
  bf16*  xp      = (bf16*) take((size_t)MROWS * EMB * 2);
  float* t       = (float*)take((size_t)MROWS * EMB * 4);
  bf16*  ybuf    = (bf16*) take((size_t)MROWS * EMB * 2);
  bf16*  hbuf    = (bf16*) take((size_t)MROWS * EMB * 2);
  bf16*  qb      = (bf16*) take((size_t)BH * NT * HD * 2);
  bf16*  kb      = (bf16*) take((size_t)BH * NT * HD * 2);
  bf16*  vt      = (bf16*) take((size_t)BH * NT * HD * 2);
  float* S       = (float*)take((size_t)BH * NT * NT * 4);
  bf16*  attn_o  = (bf16*) take((size_t)MROWS * EMB * 2);
  bf16*  h1      = (bf16*) take((size_t)MROWS * FF * 2);
  bf16*  tb      = (bf16*) take((size_t)MROWS * EMB * 2);
  bf16*  oph     = (bf16*) take((size_t)MROWS * INTR * 2);
  float* h2      = (float*)take((size_t)MROWS * EMB * 4);
  (void)in_sizes; (void)n_in; (void)out_size; (void)ws_size;

  auto cvt = [&](const float* s, bf16* d, size_t n) {
    k_cvt<<<dim3((unsigned)((n + 255) / 256)), dim3(256), 0, stream>>>(s, d, (int)n);
  };
  // weights -> bf16 (77 MB total; resident in the 192 MB L2 for the whole pass)
  cvt(conv_w, wb_conv, (size_t)EMB * EMB);
  cvt(qkv_w,  wb_qkv,  (size_t)LAY * TE * EMB);
  cvt(proj_w, wb_proj, (size_t)LAY * EMB * EMB);
  cvt(mlp_w1, wb_m1,   (size_t)LAY * FF * EMB);
  cvt(mlp_w2, wb_m2,   (size_t)LAY * EMB * FF);
  cvt(op_w1,  wb_o1,   (size_t)INTR * EMB);
  cvt(op_w2,  wb_o2,   (size_t)EMB * INTR);

  k_pe<<<dim3((NT * EMB + 255) / 256), dim3(256), 0, stream>>>(pe);
  k_im2col<<<dim3((BATCH * NT * EMB + 255) / 256), dim3(256), 0, stream>>>(x, xp);
  k_patch_gemm<<<dim3(MROWS / 16, EMB / 64), dim3(32), 0, stream>>>(xp, wb_conv, conv_b, pe, t);

  for (int l = 0; l < LAY; ++l) {
    const bf16* Wq = wb_qkv  + (size_t)l * TE * EMB;
    const bf16* Wp = wb_proj + (size_t)l * EMB * EMB;
    const bf16* W1 = wb_m1   + (size_t)l * FF * EMB;
    const bf16* W2 = wb_m2   + (size_t)l * EMB * FF;

    k_ln<<<dim3(MROWS), dim3(256), 0, stream>>>(t, ln1_g + l * EMB, ln1_b + l * EMB, ybuf);
    k_qkv_gemm<<<dim3(MROWS / 16, TE / 64), dim3(32), 0, stream>>>(ybuf, Wq, qb, kb, vt);
    k_score_gemm<<<dim3(NT / 16, NT / 64, BH), dim3(32), 0, stream>>>(
        qb, kb, attn_bi + (size_t)l * NT * NT, S);
    k_softmax<<<dim3(NT, BH), dim3(256), 0, stream>>>(S);
    k_av_gemm<<<dim3(NT / 16, 1, BH), dim3(32), 0, stream>>>(S, vt, attn_o);
    k_proj_gemm<<<dim3(MROWS / 16, EMB / 64), dim3(32), 0, stream>>>(
        attn_o, Wp, proj_b + l * EMB, ybuf, t);
    k_ln<<<dim3(MROWS), dim3(256), 0, stream>>>(t, ln2_g + l * EMB, ln2_b + l * EMB, hbuf);
    k_mlp1_gemm<<<dim3(MROWS / 16, FF / 64), dim3(32), 0, stream>>>(hbuf, W1, mlp_b1 + l * FF, h1);
    k_mlp2_gemm<<<dim3(MROWS / 16, EMB / 64), dim3(32), 0, stream>>>(h1, W2, mlp_b2 + l * EMB, t);
  }

  cvt(t, tb, (size_t)MROWS * EMB);
  k_op1_gemm<<<dim3(MROWS / 16, INTR / 64), dim3(32), 0, stream>>>(tb, wb_o1, op_b1, oph);
  k_op2_gemm<<<dim3(MROWS / 16, EMB / 64), dim3(32), 0, stream>>>(oph, wb_o2, op_b2, h2);
  k_gate<<<dim3((BATCH * IMG * IMG + 255) / 256), dim3(256), 0, stream>>>(
      h2, x, ag_wg, ag_bngw, ag_bngb, ag_wx, ag_bnxw, ag_bnxb, ag_psiw, ag_psib, out);
}
